// LSTM_26448408609158
// MI455X (gfx1250) — compile-verified, matrix operand order checked by reading, not run on hardware
//
#include <hip/hip_runtime.h>
#include <hip/hip_bf16.h>
#include <stdint.h>

// ---------------------------------------------------------------------------
// 3-layer LSTM recurrence (64 outer steps x 128 inner seq steps, strictly
// sequential) on MI455X.
//  * bf16 WMMA mat-vecs: v_wmma_f32_16x16x32_bf16, h replicated over N,
//    4 independent accumulator chains to pipeline the XDL unit.
//  * Weights live in VGPRs: each wave holds its 16x512 row-tile of both
//    matrices (256 VGPRs/lane) for the entire kernel -> no per-step weight
//    streaming (one-time 12.6 MB load instead of ~103 GB of L2 traffic).
//  * Layer pipeline: 48 persistent WGs = 3 layer-groups x 16; tick k runs
//    layer l on seq step s=k-l, one device barrier per tick.
//  * CDNA5 paths: async global->LDS staging (global_load_async_to_lds_b128 +
//    s_wait_asynccnt), s_cluster_barrier fast path, s_sleep spin.
// ---------------------------------------------------------------------------

typedef __attribute__((ext_vector_type(16))) __bf16 v16bf;
typedef __attribute__((ext_vector_type(8)))  float  v8f;

#define HIDDEN 512
#define BATCH  128           // consumed as sequence length by the inner LSTM
#define TSTEPS 64
#define NLAYER 3
#define G4H    2048          // 4*HIDDEN gate rows
#define NWG    48            // 3 layer-groups x 16 WGs
#define NTHR   256           // 8 wave32 per WG
#define KC     16            // 512 / 32 K-chunks per mat-vec
#define NTICK  (BATCH + NLAYER - 1)                 // 130 pipeline ticks
#define TILE_ELEMS (16*32*16)                       // halves per (layer,mat,row-tile)
#define PACKED_TOTAL (NLAYER*2*128*KC*32*16)        // 6,291,456 bf16

// workspace layout (bytes, 256B-aligned offsets); total ~12.9 MB
#define OFF_PACKED 0
#define OFF_BIAS   (PACKED_TOTAL*2)                 // 12,582,912
#define OFF_SEQ    (OFF_BIAS + NLAYER*G4H*4)        // +24,576
#define OFF_HBUF   (OFF_SEQ + 2*BATCH*HIDDEN*2)     // +262,144
#define OFF_SYNC   (OFF_HBUF + 2*NLAYER*HIDDEN*2)   // +6,144

// ---------------------------------------------------------------------------
// Prep: bf16-swizzle weights into this kernel's contiguous-k A-operand order (lane l of a wave owns
// row (l&15) and K sub-range (l>>4)*16 of each 32-wide K chunk -> 16
// contiguous halves per lane per WMMA), fuse biases, seed seq[0]=x[:,0,:],
// init d_out = b_lin, reset barrier state.
// ---------------------------------------------------------------------------
extern "C" __global__ void __launch_bounds__(256)
lstm_prep(const float* __restrict__ x,
          const float* __restrict__ Wih, const float* __restrict__ Whh,
          const float* __restrict__ bih, const float* __restrict__ bhh,
          const float* __restrict__ b_lin,
          __bf16* __restrict__ packedW, float* __restrict__ bias,
          __bf16* __restrict__ seq0, float* __restrict__ out,
          unsigned* __restrict__ sync)
{
  const int idx = blockIdx.x * 256 + threadIdx.x;
  if (idx < PACKED_TOTAL) {
    const int e    = idx & 15;
    const int lane = (idx >> 4) & 31;
    const int kc   = (idx >> 9) & 15;
    const int tile = (idx >> 13) & 127;
    const int mat  = (idx >> 20) & 1;
    const int l    = idx >> 21;
    const int row  = tile * 16 + (lane & 15);
    const int k    = kc * 32 + ((lane >> 4) << 4) + e;
    const float* src = mat ? Whh : Wih;
    packedW[idx] = (__bf16)src[((size_t)l * G4H + row) * HIDDEN + k];
  }
  if (idx < NLAYER * G4H) bias[idx] = bih[idx] + bhh[idx];
  if (idx < BATCH * HIDDEN) {
    const int b = idx >> 9, j = idx & 511;
    seq0[idx] = (__bf16)x[((size_t)b * TSTEPS + 0) * HIDDEN + j];
  }
  if (idx < BATCH * TSTEPS) out[idx] = b_lin[0];
  if (idx < 2) sync[idx * 16] = 0u;   // cnt @ [0], gen @ [16]
}

// ---------------------------------------------------------------------------
// Device-scope sense-reversal barrier across the NWG persistent workgroups.
// s_cluster_barrier is the HW fast path under cluster dispatch and an
// architectural NOP (ClusterID==0) otherwise; the atomic path is the
// correctness guarantee.
// ---------------------------------------------------------------------------
__device__ __forceinline__ void grid_sync(unsigned* cnt, unsigned* gen)
{
  __syncthreads();
  if (threadIdx.x == 0) {
    __threadfence();
    unsigned g = __hip_atomic_load(gen, __ATOMIC_RELAXED, __HIP_MEMORY_SCOPE_AGENT);
    if (__hip_atomic_fetch_add(cnt, 1u, __ATOMIC_ACQ_REL, __HIP_MEMORY_SCOPE_AGENT) == NWG - 1u) {
      __hip_atomic_store(cnt, 0u, __ATOMIC_RELAXED, __HIP_MEMORY_SCOPE_AGENT);
      __hip_atomic_store(gen, g + 1u, __ATOMIC_RELEASE, __HIP_MEMORY_SCOPE_AGENT);
    } else {
      while (__hip_atomic_load(gen, __ATOMIC_ACQUIRE, __HIP_MEMORY_SCOPE_AGENT) == g)
        __builtin_amdgcn_s_sleep(1);
    }
  }
  __builtin_amdgcn_s_cluster_barrier();
  __syncthreads();
  __threadfence();
}

// ---------------------------------------------------------------------------
// Persistent pipelined kernel. WG b: layer l = b>>4, hidden slice w = b&15
// (owns hidden lanes [32w,32w+32) of its layer; c-state never leaves LDS).
// Wave v owns gate-row tile (v>>1)*32 + 2w + (v&1); its weights stay in VGPRs.
// ---------------------------------------------------------------------------
extern "C" __global__ void __launch_bounds__(NTHR, 2)
lstm_persistent(const __bf16* __restrict__ packedW,
                const float*  __restrict__ bias,
                const float*  __restrict__ x,
                const float*  __restrict__ w_lin,
                __bf16* __restrict__ seqbuf,
                __bf16* __restrict__ hbuf,     // [NLAYER][2 parities][512] bf16
                float*  __restrict__ out,
                unsigned* cnt, unsigned* gen)
{
  const int l    = blockIdx.x >> 4;     // layer group 0..2
  const int w    = blockIdx.x & 15;     // hidden slice 0..15
  const int tid  = threadIdx.x;
  const int lane = tid & 31;
  const int wave = tid >> 5;
  const int tile = ((wave >> 1) * 32) + (w * 2) + (wave & 1);  // row-tile 0..127
  const int J0   = w * 32;

  __shared__ __attribute__((aligned(32))) __bf16 vIn[HIDDEN];
  __shared__ __attribute__((aligned(32))) __bf16 vHid[HIDDEN];
  __shared__ float gateBuf[128];
  __shared__ float biasLds[128];
  __shared__ float cState[32];

  for (int i = tid; i < 128; i += NTHR) {
    const int g = i >> 5, ww = i & 31;
    biasLds[i] = bias[l * G4H + g * HIDDEN + J0 + ww];
  }

  // ---- one-time: this wave's weight tile -> 256 VGPRs/lane ----------------
  v16bf wIh[KC], wHh[KC];
  {
    const __bf16* gI = packedW + (size_t)((l * 2 + 0) * 128 + tile) * TILE_ELEMS + (size_t)lane * 16;
    const __bf16* gH = packedW + (size_t)((l * 2 + 1) * 128 + tile) * TILE_ELEMS + (size_t)lane * 16;
    #pragma unroll
    for (int kc = 0; kc < KC; ++kc) {
      wIh[kc] = *(const v16bf*)(gI + kc * (32 * 16));
      wHh[kc] = *(const v16bf*)(gH + kc * (32 * 16));
    }
  }
  __syncthreads();

  for (int t = 0; t < TSTEPS; ++t) {
    const __bf16* seqIn  = seqbuf + (size_t)(t & 1) * (BATCH * HIDDEN);
    __bf16*       seqOut = seqbuf + (size_t)((t + 1) & 1) * (BATCH * HIDDEN);

    // fresh zero LSTM state each outer timestep (reference semantics)
    if (tid < 32) {
      cState[tid] = 0.0f;
      hbuf[(l * 2 + 0) * HIDDEN + J0 + tid] = (__bf16)0.0f;
      hbuf[(l * 2 + 1) * HIDDEN + J0 + tid] = (__bf16)0.0f;
    }
    grid_sync(cnt, gen);

    for (int k = 0; k < NTICK; ++k) {
      const int s = k - l;                   // this layer's seq step this tick
      if (s >= 0 && s < BATCH) {             // uniform per-WG predicate
        const int pr = (k + 1) & 1;          // parity written at tick k-1
        const int pw = k & 1;
        const __bf16* inp = (l == 0) ? (seqIn + (size_t)s * HIDDEN)
                                     : (hbuf + ((l - 1) * 2 + pr) * HIDDEN);
        const __bf16* hid = hbuf + (l * 2 + pr) * HIDDEN;

        // stage both 512-wide bf16 vectors via the CDNA5 async LDS path
        // (waves 0..3: 128 lanes x 16B, b128 flavor)
        if (tid < 128) {
          const __bf16* src  = (tid < 64) ? (inp + tid * 8) : (hid + (tid - 64) * 8);
          __bf16*       dstp = (tid < 64) ? (vIn + tid * 8) : (vHid + (tid - 64) * 8);
          unsigned ldsoff = (unsigned)(uintptr_t)dstp;
          asm volatile("global_load_async_to_lds_b128 %0, %1, off"
                       :: "v"(ldsoff), "v"(src) : "memory");
          asm volatile("s_wait_asynccnt 0" ::: "memory");
        }
        __syncthreads();

        // 4 independent WMMA accumulation chains (Wih/Whh x even/odd kc) so
        // the XDL pipe can issue without back-to-back C->D dependencies.
        v8f a0 = {0.f,0.f,0.f,0.f,0.f,0.f,0.f,0.f};
        v8f a1 = a0, a2 = a0, a3 = a0;
        const int kseg = (lane >> 4) << 4;   // this lane's K sub-range (0 or 16)
        #pragma unroll
        for (int kp = 0; kp < KC / 2; ++kp) {
          const int k0 = 2 * kp, k1 = 2 * kp + 1;
          v16bf bI0 = *(const v16bf*)(vIn  + k0 * 32 + kseg);
          v16bf bI1 = *(const v16bf*)(vIn  + k1 * 32 + kseg);
          v16bf bH0 = *(const v16bf*)(vHid + k0 * 32 + kseg);
          v16bf bH1 = *(const v16bf*)(vHid + k1 * 32 + kseg);
          a0 = __builtin_amdgcn_wmma_f32_16x16x32_bf16(false, wIh[k0], false, bI0,
                                                       (short)0, a0, false, false);
          a1 = __builtin_amdgcn_wmma_f32_16x16x32_bf16(false, wIh[k1], false, bI1,
                                                       (short)0, a1, false, false);
          a2 = __builtin_amdgcn_wmma_f32_16x16x32_bf16(false, wHh[k0], false, bH0,
                                                       (short)0, a2, false, false);
          a3 = __builtin_amdgcn_wmma_f32_16x16x32_bf16(false, wHh[k1], false, bH1,
                                                       (short)0, a3, false, false);
        }
        const v8f acc = (a0 + a1) + (a2 + a3);

        // D column 0: lane 0 -> M=0..7 (VGPR 0..7), lane 16 -> M=8..15
        if (lane == 0 || lane == 16) {
          const int mbase = (lane >> 4) * 8;
          #pragma unroll
          for (int q = 0; q < 8; ++q)
            gateBuf[wave * 16 + mbase + q] =
                acc[q] + biasLds[wave * 16 + mbase + q];
        }
        __syncthreads();

        if (tid < 32) {
          const int jl = tid;
          float gi = gateBuf[jl], gf = gateBuf[32 + jl];
          float gg = gateBuf[64 + jl], go = gateBuf[96 + jl];
          gi = 1.0f / (1.0f + __expf(-gi));
          gf = 1.0f / (1.0f + __expf(-gf));
          gg = tanhf(gg);
          go = 1.0f / (1.0f + __expf(-go));
          const float c = gf * cState[jl] + gi * gg;
          cState[jl] = c;
          const float h = go * tanhf(c);
          hbuf[(l * 2 + pw) * HIDDEN + J0 + jl] = (__bf16)h;
          if (l == NLAYER - 1) {
            // teacher-forcing blend (tf=0.5) + fused final Linear
            const float xv = x[((size_t)s * TSTEPS + t) * HIDDEN + J0 + jl];
            seqOut[(size_t)s * HIDDEN + J0 + jl] = (__bf16)(0.5f * (h + xv));
            float pd = w_lin[J0 + jl] * h;
            #pragma unroll
            for (int off = 16; off > 0; off >>= 1) pd += __shfl_down(pd, off, 32);
            if (jl == 0) atomicAdd(&out[s * TSTEPS + t], pd);
          }
        }
      }
      grid_sync(cnt, gen);   // one device barrier per pipeline tick
    }
  }
}

// ---------------------------------------------------------------------------
extern "C" void kernel_launch(void* const* d_in, const int* in_sizes, int n_in,
                              void* d_out, int out_size, void* d_ws, size_t ws_size,
                              hipStream_t stream)
{
  (void)in_sizes; (void)n_in; (void)out_size; (void)ws_size;  // ws needs ~12.9 MB
  const float* x    = (const float*)d_in[0];
  const float* Wih  = (const float*)d_in[1];
  const float* Whh  = (const float*)d_in[2];
  const float* bih  = (const float*)d_in[3];
  const float* bhh  = (const float*)d_in[4];
  const float* wlin = (const float*)d_in[5];
  const float* blin = (const float*)d_in[6];

  char* ws = (char*)d_ws;
  __bf16*   packedW = (__bf16*)(ws + OFF_PACKED);
  float*    bias    = (float*)(ws + OFF_BIAS);
  __bf16*   seqbuf  = (__bf16*)(ws + OFF_SEQ);
  __bf16*   hbuf    = (__bf16*)(ws + OFF_HBUF);
  unsigned* syncw   = (unsigned*)(ws + OFF_SYNC);
  float*    out     = (float*)d_out;

  const int prepBlocks = (PACKED_TOTAL + 255) / 256;
  lstm_prep<<<prepBlocks, 256, 0, stream>>>(x, Wih, Whh, bih, bhh, blin,
                                            packedW, bias, seqbuf, out, syncw);
  lstm_persistent<<<NWG, NTHR, 0, stream>>>(packedW, bias, x, wlin,
                                            seqbuf, hbuf, out,
                                            &syncw[0], &syncw[16]);
}
